// SequentialTransductionUnitDense_68934225101348
// MI455X (gfx1250) — matrix operand order repaired and checked by reading, not compiled, not run on hardware
//
#include <hip/hip_runtime.h>
#include <hip/hip_bf16.h>
#include <cstdint>

// HSTU dense forward for MI455X (gfx1250), wave32, WMMA bf16 path,
// async global->LDS double buffering (ASYNCcnt) in the big GEMMs.
// B=4, N=2048, D=512, H=8, dA=dL=64.

#define NSEQ   2048
#define DMODEL 512
#define EPROJ  2048
#define HEADS  8

typedef __attribute__((ext_vector_type(16))) __bf16 bf16x16;
typedef __attribute__((ext_vector_type(8)))  float  f32x8;

__device__ __forceinline__ unsigned short f2bf(float f) {
  unsigned int u = __builtin_bit_cast(unsigned int, f);
  u += 0x7FFFu + ((u >> 16) & 1u);   // round-to-nearest-even
  return (unsigned short)(u >> 16);
}

__device__ __forceinline__ float silu_f(float x) {
  return x / (1.0f + __expf(-x));
}

struct Bytes32 { uint4 lo, hi; };

// A fragment: 16x32 bf16, LDS tile row-major [m][k] with row stride ldk.
// lane L: m = L%16, h = L/16; elems 0..7 -> K=8h..8h+7, elems 8..15 -> K=16+8h..+7
__device__ __forceinline__ bf16x16 frag_a(const unsigned short* base, int ldk, int kk) {
  int lane = threadIdx.x & 31;
  int m = lane & 15, h = lane >> 4;
  const unsigned short* p = base + m * ldk + kk + 8 * h;
  Bytes32 f;
  f.lo = *(const uint4*)(p);
  f.hi = *(const uint4*)(p + 16);
  return __builtin_bit_cast(bf16x16, f);
}

// B fragment: 32x16 bf16 stored transposed in LDS as [n][k], row stride ldk.
// lane L: n = L%16, h = L/16; elems 0..7 -> K=8h..8h+7, elems 8..15 -> K=16+8h..+7
__device__ __forceinline__ bf16x16 frag_b(const unsigned short* base, int ldk, int kk) {
  int lane = threadIdx.x & 31;
  int n = lane & 15, h = lane >> 4;
  const unsigned short* p = base + n * ldk + kk + 8 * h;
  Bytes32 f;
  f.lo = *(const uint4*)(p);
  f.hi = *(const uint4*)(p + 16);
  return __builtin_bit_cast(bf16x16, f);
}

__device__ __forceinline__ f32x8 wmma_bf16(bf16x16 a, bf16x16 b, f32x8 c) {
  return __builtin_amdgcn_wmma_f32_16x16x32_bf16(false, a, false, b, (short)0, c,
                                                 false, false);
}

// Async global->LDS copy of one 128x64 bf16 A tile + one 128x64 B tile.
// Each of 256 threads issues 8 GLOBAL_LOAD_ASYNC_TO_LDS_B128 (tracked by ASYNCcnt).
__device__ __forceinline__ void async_tile_issue(
    unsigned short* As, unsigned short* Bs,
    const unsigned short* ga, const unsigned short* gb, int t) {
#pragma unroll
  for (int ii = 0; ii < 4; ++ii) {
    int i = t + ii * 256;
    int r = i >> 3, c = (i & 7) << 3;
    asm volatile("global_load_async_to_lds_b128 %0, %1, off"
                 :: "v"((unsigned int)(uintptr_t)(As + r * 64 + c)),
                    "v"(ga + (size_t)r * DMODEL + c)
                 : "memory");
    asm volatile("global_load_async_to_lds_b128 %0, %1, off"
                 :: "v"((unsigned int)(uintptr_t)(Bs + r * 64 + c)),
                    "v"(gb + (size_t)r * DMODEL + c)
                 : "memory");
  }
}

// ---------------- pack kernels ----------------
__global__ void k_pack_uvqk(const float* __restrict__ w, unsigned short* __restrict__ wt) {
  int idx = blockIdx.x * blockDim.x + threadIdx.x;   // col*512 + k
  int k = idx & (DMODEL - 1), col = idx >> 9;
  wt[idx] = f2bf(w[k * EPROJ + col]);
}

__global__ void k_pack_ow(const float* __restrict__ w, unsigned short* __restrict__ wb) {
  int idx = blockIdx.x * blockDim.x + threadIdx.x;
  wb[idx] = f2bf(w[idx]);
}

// ---------------- layernorm(x) -> bf16 ----------------
__global__ void k_ln_x(const float* __restrict__ x, unsigned short* __restrict__ nx) {
  int row = blockIdx.x, t = threadIdx.x;
  const float* xr = x + (size_t)row * DMODEL;
  float v0 = xr[t], v1 = xr[t + 256];
  __shared__ float red[256];
  red[t] = v0 + v1; __syncthreads();
  for (int s = 128; s > 0; s >>= 1) { if (t < s) red[t] += red[t + s]; __syncthreads(); }
  float mean = red[0] * (1.0f / DMODEL);
  __syncthreads();
  float d0 = v0 - mean, d1 = v1 - mean;
  red[t] = d0 * d0 + d1 * d1; __syncthreads();
  for (int s = 128; s > 0; s >>= 1) { if (t < s) red[t] += red[t + s]; __syncthreads(); }
  float rstd = rsqrtf(red[0] * (1.0f / DMODEL) + 1e-6f);
  nx[(size_t)row * DMODEL + t]       = f2bf(d0 * rstd);
  nx[(size_t)row * DMODEL + t + 256] = f2bf(d1 * rstd);
}

// ---------------- projection GEMM + SiLU ----------------
// C[8192 x 2048] = normed_x @ uvqk_t. 128x128/WG, 32x64/wave, async dbl-buffered LDS.
__global__ __launch_bounds__(256) void k_proj(
    const unsigned short* __restrict__ nx,   // [8192][512]
    const unsigned short* __restrict__ wt,   // [2048][512]  ([col][k])
    float* __restrict__ u,
    unsigned short* __restrict__ vbuf,
    unsigned short* __restrict__ qbuf,
    unsigned short* __restrict__ kbuf) {
  __shared__ __align__(16) unsigned short As[2][128 * 64];
  __shared__ __align__(16) unsigned short Bs[2][128 * 64];
  int m0 = blockIdx.x * 128, n0 = blockIdx.y * 128;
  int t = threadIdx.x, w = t >> 5, lane = t & 31;
  int mrow = (w >> 1) * 32, ncol = (w & 1) * 64;
  const unsigned short* ga = nx + (size_t)m0 * DMODEL;
  const unsigned short* gb = wt + (size_t)n0 * DMODEL;

  f32x8 acc[8];
#pragma unroll
  for (int j = 0; j < 8; ++j) acc[j] = (f32x8)(0.0f);

  async_tile_issue(As[0], Bs[0], ga, gb, t);
  for (int k0 = 0; k0 < DMODEL; k0 += 64) {
    int p = (k0 >> 6) & 1;
    if (k0 + 64 < DMODEL) {
      async_tile_issue(As[p ^ 1], Bs[p ^ 1], ga + k0 + 64, gb + k0 + 64, t);
      asm volatile("s_wait_asynccnt 0x8" ::: "memory");  // prev tile's 8 ops done
    } else {
      asm volatile("s_wait_asynccnt 0x0" ::: "memory");
    }
    __syncthreads();
    const unsigned short* Ap = As[p];
    const unsigned short* Bp = Bs[p];
#pragma unroll
    for (int kk = 0; kk < 64; kk += 32) {
      bf16x16 af0 = frag_a(Ap + mrow * 64, 64, kk);
      bf16x16 af1 = frag_a(Ap + (mrow + 16) * 64, 64, kk);
      bf16x16 bfr[4];
#pragma unroll
      for (int j = 0; j < 4; ++j) bfr[j] = frag_b(Bp + (ncol + j * 16) * 64, 64, kk);
#pragma unroll
      for (int j = 0; j < 4; ++j) acc[j]     = wmma_bf16(af0, bfr[j], acc[j]);
#pragma unroll
      for (int j = 0; j < 4; ++j) acc[4 + j] = wmma_bf16(af1, bfr[j], acc[4 + j]);
    }
    __syncthreads();
  }

  int n = lane & 15, h = lane >> 4;
#pragma unroll
  for (int i = 0; i < 2; ++i) {
    int mbase = m0 + mrow + i * 16 + h * 8;
#pragma unroll
    for (int j = 0; j < 4; ++j) {
      int col = n0 + ncol + j * 16 + n;
#pragma unroll
      for (int r = 0; r < 8; ++r) {
        float s = silu_f(acc[i * 4 + j][r]);
        size_t row = (size_t)(mbase + r);
        if (col < 512)       u[row * 512 + col]             = s;
        else if (col < 1024) vbuf[row * 512 + (col - 512)]  = f2bf(s);
        else if (col < 1536) qbuf[row * 512 + (col - 1024)] = f2bf(s);
        else                 kbuf[row * 512 + (col - 1536)] = f2bf(s);
      }
    }
  }
}

// ---------------- HSTU attention (causal, silu(S)/N, no softmax) ----------------
__global__ __launch_bounds__(256) void k_attn(
    const unsigned short* __restrict__ qbuf,
    const unsigned short* __restrict__ kbuf,
    const unsigned short* __restrict__ vbuf,
    float* __restrict__ attn) {
  __shared__ __align__(16) unsigned short Qs[128 * 64];
  __shared__ __align__(16) unsigned short Ks[64 * 64];
  __shared__ __align__(16) unsigned short Vts[64 * 64];  // [feature][key]
  __shared__ __align__(16) unsigned short Ss[128 * 64];
  int t = threadIdx.x, w = t >> 5, lane = t & 31;
  int n = lane & 15, h = lane >> 4;
  int q0 = blockIdx.x * 128;
  int b = blockIdx.y >> 3, hh = blockIdx.y & 7;
  const unsigned short* qb = qbuf + (size_t)b * NSEQ * DMODEL + hh * 64;
  const unsigned short* kb = kbuf + (size_t)b * NSEQ * DMODEL + hh * 64;
  const unsigned short* vb = vbuf + (size_t)b * NSEQ * DMODEL + hh * 64;

  for (int i = t; i < 1024; i += 256) {              // Q tile once
    int r = i >> 3, c = (i & 7) << 3;
    *(uint4*)(Qs + r * 64 + c) = *(const uint4*)(qb + (size_t)(q0 + r) * DMODEL + c);
  }
  __syncthreads();
  // Q fragments are loop-invariant: hoist out of the key loop.
  bf16x16 qa0 = frag_a(Qs + (w * 16) * 64, 64, 0);
  bf16x16 qa1 = frag_a(Qs + (w * 16) * 64, 64, 32);

  f32x8 accO[4];
#pragma unroll
  for (int j = 0; j < 4; ++j) accO[j] = (f32x8)(0.0f);
  const float invN = 1.0f / (float)NSEQ;

  for (int kt = 0; kt <= q0 + 127; kt += 64) {       // causal: skip future tiles
    __syncthreads();
    for (int i = t; i < 512; i += 256) {             // 64 rows x 8 uint4
      int r = i >> 3, c = (i & 7) << 3;
      *(uint4*)(Ks + r * 64 + c) = *(const uint4*)(kb + (size_t)(kt + r) * DMODEL + c);
      uint4 vv = *(const uint4*)(vb + (size_t)(kt + r) * DMODEL + c);
      const unsigned short* ve = (const unsigned short*)&vv;
#pragma unroll
      for (int e = 0; e < 8; ++e) Vts[(c + e) * 64 + r] = ve[e];  // transpose into LDS
    }
    __syncthreads();

    f32x8 accS[4];
#pragma unroll
    for (int j = 0; j < 4; ++j) accS[j] = (f32x8)(0.0f);
    {
      bf16x16 fr[4];
#pragma unroll
      for (int j = 0; j < 4; ++j) fr[j] = frag_b(Ks + (j * 16) * 64, 64, 0);
#pragma unroll
      for (int j = 0; j < 4; ++j) accS[j] = wmma_bf16(qa0, fr[j], accS[j]);
#pragma unroll
      for (int j = 0; j < 4; ++j) fr[j] = frag_b(Ks + (j * 16) * 64, 64, 32);
#pragma unroll
      for (int j = 0; j < 4; ++j) accS[j] = wmma_bf16(qa1, fr[j], accS[j]);
    }

    // silu(S)/N + analytic causal mask; each wave writes only its own 16 rows
    int qrow = q0 + w * 16 + h * 8;
#pragma unroll
    for (int j = 0; j < 4; ++j) {
      int ki = kt + j * 16 + n;
#pragma unroll
      for (int r = 0; r < 8; ++r) {
        float s = (ki <= qrow + r) ? silu_f(accS[j][r]) * invN : 0.0f;
        Ss[(w * 16 + h * 8 + r) * 64 + j * 16 + n] = f2bf(s);
      }
    }

    // O += S @ V  (A = own rows of Ss -> intra-wave DS ordering, no barrier)
#pragma unroll
    for (int kk = 0; kk < 64; kk += 32) {
      bf16x16 a = frag_a(Ss + (w * 16) * 64, 64, kk);
      bf16x16 fr[4];
#pragma unroll
      for (int j = 0; j < 4; ++j) fr[j] = frag_b(Vts + (j * 16) * 64, 64, kk);
#pragma unroll
      for (int j = 0; j < 4; ++j) accO[j] = wmma_bf16(a, fr[j], accO[j]);
    }
  }

  float* ao = attn + ((size_t)b * NSEQ + q0 + w * 16 + h * 8) * DMODEL + hh * 64;
#pragma unroll
  for (int j = 0; j < 4; ++j)
#pragma unroll
    for (int r = 0; r < 8; ++r)
      ao[(size_t)r * DMODEL + j * 16 + n] = accO[j][r];
}

// ---------------- layernorm(attn) * u -> bf16 ----------------
__global__ void k_lnmul(const float* __restrict__ attn, const float* __restrict__ u,
                        unsigned short* __restrict__ oi) {
  int row = blockIdx.x, t = threadIdx.x;
  const float* ar = attn + (size_t)row * DMODEL;
  float v0 = ar[t], v1 = ar[t + 256];
  __shared__ float red[256];
  red[t] = v0 + v1; __syncthreads();
  for (int s = 128; s > 0; s >>= 1) { if (t < s) red[t] += red[t + s]; __syncthreads(); }
  float mean = red[0] * (1.0f / DMODEL);
  __syncthreads();
  float d0 = v0 - mean, d1 = v1 - mean;
  red[t] = d0 * d0 + d1 * d1; __syncthreads();
  for (int s = 128; s > 0; s >>= 1) { if (t < s) red[t] += red[t + s]; __syncthreads(); }
  float rstd = rsqrtf(red[0] * (1.0f / DMODEL) + 1e-6f);
  const float* ur = u + (size_t)row * DMODEL;
  oi[(size_t)row * DMODEL + t]       = f2bf(d0 * rstd * ur[t]);
  oi[(size_t)row * DMODEL + t + 256] = f2bf(d1 * rstd * ur[t + 256]);
}

// ---------------- output GEMM + bias + residual ----------------
__global__ __launch_bounds__(256) void k_out(
    const unsigned short* __restrict__ oi,   // [8192][512]
    const unsigned short* __restrict__ owb,  // [512][512] ([n=d][k=f])
    const float* __restrict__ ob,
    const float* __restrict__ x,
    float* __restrict__ out) {
  __shared__ __align__(16) unsigned short As[2][128 * 64];
  __shared__ __align__(16) unsigned short Bs[2][128 * 64];
  int m0 = blockIdx.x * 128, n0 = blockIdx.y * 128;
  int t = threadIdx.x, w = t >> 5, lane = t & 31;
  int mrow = (w >> 1) * 32, ncol = (w & 1) * 64;
  const unsigned short* ga = oi  + (size_t)m0 * DMODEL;
  const unsigned short* gb = owb + (size_t)n0 * DMODEL;

  f32x8 acc[8];
#pragma unroll
  for (int j = 0; j < 8; ++j) acc[j] = (f32x8)(0.0f);

  async_tile_issue(As[0], Bs[0], ga, gb, t);
  for (int k0 = 0; k0 < DMODEL; k0 += 64) {
    int p = (k0 >> 6) & 1;
    if (k0 + 64 < DMODEL) {
      async_tile_issue(As[p ^ 1], Bs[p ^ 1], ga + k0 + 64, gb + k0 + 64, t);
      asm volatile("s_wait_asynccnt 0x8" ::: "memory");
    } else {
      asm volatile("s_wait_asynccnt 0x0" ::: "memory");
    }
    __syncthreads();
    const unsigned short* Ap = As[p];
    const unsigned short* Bp = Bs[p];
#pragma unroll
    for (int kk = 0; kk < 64; kk += 32) {
      bf16x16 af0 = frag_a(Ap + mrow * 64, 64, kk);
      bf16x16 af1 = frag_a(Ap + (mrow + 16) * 64, 64, kk);
      bf16x16 bfr[4];
#pragma unroll
      for (int j = 0; j < 4; ++j) bfr[j] = frag_b(Bp + (ncol + j * 16) * 64, 64, kk);
#pragma unroll
      for (int j = 0; j < 4; ++j) acc[j]     = wmma_bf16(af0, bfr[j], acc[j]);
#pragma unroll
      for (int j = 0; j < 4; ++j) acc[4 + j] = wmma_bf16(af1, bfr[j], acc[4 + j]);
    }
    __syncthreads();
  }

  int n = lane & 15, h = lane >> 4;
#pragma unroll
  for (int i = 0; i < 2; ++i) {
    int mbase = m0 + mrow + i * 16 + h * 8;
#pragma unroll
    for (int j = 0; j < 4; ++j) {
      int col = n0 + ncol + j * 16 + n;
#pragma unroll
      for (int r = 0; r < 8; ++r) {
        size_t idx = (size_t)(mbase + r) * DMODEL + col;
        out[idx] = acc[i * 4 + j][r] + ob[col] + x[idx];
      }
    }
  }
}

extern "C" void kernel_launch(void* const* d_in, const int* in_sizes, int n_in,
                              void* d_out, int out_size, void* d_ws, size_t ws_size,
                              hipStream_t stream) {
  (void)in_sizes; (void)n_in; (void)out_size; (void)ws_size;
  const float* x    = (const float*)d_in[0];
  // d_in[1] = invalid_attn_mask: causal tril, applied analytically (never read)
  const float* uvqk = (const float*)d_in[2];
  const float* ow   = (const float*)d_in[3];
  const float* ob   = (const float*)d_in[4];
  float* out = (float*)d_out;

  const size_t ROWS = 4ull * NSEQ;             // 8192
  char* ws = (char*)d_ws;
  size_t off = 0;
  auto alloc = [&](size_t bytes) -> void* {
    void* p = ws + off;
    off += (bytes + 255) & ~(size_t)255;
    return p;
  };
  unsigned short* nx    = (unsigned short*)alloc(ROWS * DMODEL * 2);
  unsigned short* wt    = (unsigned short*)alloc((size_t)EPROJ * DMODEL * 2);
  unsigned short* owb   = (unsigned short*)alloc((size_t)DMODEL * DMODEL * 2);
  float*          ubuf  = (float*)alloc(ROWS * DMODEL * 4);
  unsigned short* vbuf  = (unsigned short*)alloc(ROWS * DMODEL * 2);
  unsigned short* qbuf  = (unsigned short*)alloc(ROWS * DMODEL * 2);
  unsigned short* kbuf  = (unsigned short*)alloc(ROWS * DMODEL * 2);
  float*          attn  = (float*)alloc(ROWS * DMODEL * 4);
  unsigned short* oibuf = (unsigned short*)alloc(ROWS * DMODEL * 2);

  k_pack_uvqk<<<(EPROJ * DMODEL) / 256, 256, 0, stream>>>(uvqk, wt);
  k_pack_ow<<<(DMODEL * DMODEL) / 256, 256, 0, stream>>>(ow, owb);
  k_ln_x<<<(int)ROWS, 256, 0, stream>>>(x, nx);
  k_proj<<<dim3(ROWS / 128, EPROJ / 128), 256, 0, stream>>>(nx, wt, ubuf, vbuf, qbuf, kbuf);
  k_attn<<<dim3(NSEQ / 128, 4 * HEADS), 256, 0, stream>>>(qbuf, kbuf, vbuf, attn);
  k_lnmul<<<(int)ROWS, 256, 0, stream>>>(attn, ubuf, oibuf);
  k_out<<<dim3(ROWS / 128, DMODEL / 128), 256, 0, stream>>>(oibuf, owb, ob, x, out);
}
